// CorrectLowRankSparse2to4Linear_53738630807946
// MI455X (gfx1250) — matrix operand order repaired and checked by reading, not compiled, not run on hardware
//
#include <hip/hip_runtime.h>

typedef __attribute__((ext_vector_type(16))) _Float16 v16h;
typedef __attribute__((ext_vector_type(8)))  _Float16 v8h;
typedef __attribute__((ext_vector_type(8)))  float    v8f;
typedef __attribute__((ext_vector_type(4)))  float    v4f;
typedef __attribute__((ext_vector_type(4)))  int      v4i;

// pointer types matching the async-to-LDS builtin prototype:
// (v4i addrspace(1)*, v4i addrspace(3)*, imm offset, imm cpol)
typedef v4i __attribute__((address_space(1)))* gv4i_p;
typedef v4i __attribute__((address_space(3)))* lv4i_p;

#define N_ROWS  16384   // B*S
#define IN_F    4096
#define OUT_F   4096
#define RANK    64
#define KSPLIT  4

// ---------------------------------------------------------------------------
// Fragment mappings for V_WMMA_F32_16X16X32_F16 (see CDNA5 ISA):
// A (16x32, 16-bit): lane = m + 16*hi ; element j holds K with
//   hi = (koff>>3)&1 ; j = (koff&7) + 8*(koff>>4)
// B (32x16, 16-bit): same K order as A:
//   hi = (koff>>3)&1 ; j = (koff&7) + 8*(koff>>4) ; lane = n + 16*hi
// C/D (16x16 f32): VGPR v, lanes<16 -> M=v ; lanes>=16 -> M=v+8 ; N = lane&15
// ---------------------------------------------------------------------------

// ---- gfx1250 async global->LDS copy helpers -------------------------------
__device__ __forceinline__ void async_cp16(const _Float16* g, _Float16* l) {
#if __has_builtin(__builtin_amdgcn_global_load_async_to_lds_b128)
    __builtin_amdgcn_global_load_async_to_lds_b128((gv4i_p)g, (lv4i_p)l, 0, 0);
#else
    *(v8h*)l = *(const v8h*)g;   // fallback: vmem load + ds store
#endif
}

__device__ __forceinline__ void wait_async0() {
#if __has_builtin(__builtin_amdgcn_s_wait_asynccnt)
    __builtin_amdgcn_s_wait_asynccnt(0);
#else
    __asm__ volatile("s_wait_asynccnt 0x0" ::: "memory");
#endif
}

// Stage 32 KB (16384 halves) of pre-packed fragments, 256 threads x 8 x 16B.
__device__ __forceinline__ void stage32k(const _Float16* __restrict__ g,
                                         _Float16* l, int tid) {
    #pragma unroll
    for (int i = 0; i < 8; ++i) {
        int off = (i * 256 + tid) * 8;
        async_cp16(g + off, l + off);
    }
}

// ---------------------------------------------------------------------------
// soft_threshold24 on weight_A [IN_F, RANK] (groups of 4 along RANK), scale,
// cast fp16, scatter into B-fragment layout for the down-projection:
// wAfrag[(kb*4 + ntile)*512 + lane*16 + j],  kb = K-block of 32 rows of wA.
__global__ void prep_wA_kernel(const float* __restrict__ wA,
                               const float* __restrict__ scaleA,
                               _Float16* __restrict__ wAfrag) {
    int gid = blockIdx.x * blockDim.x + threadIdx.x;     // IN_F * RANK/4
    if (gid >= IN_F * (RANK / 4)) return;
    int row = gid >> 4;          // K index (0..4095)
    int g   = gid & 15;          // group along rank
    v4f w = *(const v4f*)(wA + (size_t)row * RANK + g * 4);
    float a0 = fabsf(w[0]), a1 = fabsf(w[1]), a2 = fabsf(w[2]), a3 = fabsf(w[3]);
    float mn01 = fminf(a0, a1), mx01 = fmaxf(a0, a1);
    float mn23 = fminf(a2, a3), mx23 = fmaxf(a2, a3);
    float t = fminf(fmaxf(mn01, mn23), fminf(mx01, mx23));   // 2nd smallest
    float s = scaleA[0];
    float av[4] = {a0, a1, a2, a3};
    int kb = row >> 5, koff = row & 31;
    int hi = (koff >> 3) & 1, j = (koff & 7) + 8 * (koff >> 4); // B-frag mapping
    #pragma unroll
    for (int i = 0; i < 4; ++i) {
        float r = copysignf(fmaxf(av[i] - t, 0.0f), w[i]) * s;
        int col = g * 4 + i;
        int lane = (col & 15) + (hi << 4);
        int nt = col >> 4;                                   // 0..3
        wAfrag[(size_t)((kb * 4 + nt) * 32 + lane) * 16 + j] = (_Float16)r;
    }
}

// soft_threshold24 on weight_B [OUT_F, RANK] (groups of 4 along RANK = K of
// GEMM2), scale, cast fp16, scatter into B-fragment layout of wB^T:
// wBfrag[(ntile*2 + kb)*512 + lane*16 + j], ntile over OUT_F/16, kb over 2.
__global__ void prep_wB_kernel(const float* __restrict__ wB,
                               const float* __restrict__ scaleB,
                               _Float16* __restrict__ wBfrag) {
    int gid = blockIdx.x * blockDim.x + threadIdx.x;     // OUT_F * RANK/4
    if (gid >= OUT_F * (RANK / 4)) return;
    int o = gid >> 4;            // output feature (N column of GEMM2)
    int g = gid & 15;            // group along rank (K of GEMM2)
    v4f w = *(const v4f*)(wB + (size_t)o * RANK + g * 4);
    float a0 = fabsf(w[0]), a1 = fabsf(w[1]), a2 = fabsf(w[2]), a3 = fabsf(w[3]);
    float mn01 = fminf(a0, a1), mx01 = fmaxf(a0, a1);
    float mn23 = fminf(a2, a3), mx23 = fmaxf(a2, a3);
    float t = fminf(fmaxf(mn01, mn23), fminf(mx01, mx23));
    float s = scaleB[0];
    float av[4] = {a0, a1, a2, a3};
    int nt = o >> 4;
    int lanelo = o & 15;
    #pragma unroll
    for (int i = 0; i < 4; ++i) {
        float r = copysignf(fmaxf(av[i] - t, 0.0f), w[i]) * s;
        int K = g * 4 + i;
        int kb = K >> 5, koff = K & 31;
        int hi = (koff >> 3) & 1, j = (koff & 7) + 8 * (koff >> 4); // B-frag mapping
        int lane = lanelo + (hi << 4);
        wBfrag[(size_t)((nt * 2 + kb) * 32 + lane) * 16 + j] = (_Float16)r;
    }
}

// ---------------------------------------------------------------------------
// GEMM1: x_proj_partial[ks][N, 64] = fp16(x[N, k-slice]) @ wA16[k-slice, 64]
// Block = 8 waves = 128 rows; K range 1024 (split 4 ways across blockIdx.y).
// Weight fragments double-buffered in LDS via async global->LDS copies.
__global__ void __launch_bounds__(256)
lora_down_kernel(const float* __restrict__ x,
                 const _Float16* __restrict__ wAfrag,
                 float* __restrict__ xpart) {
    __shared__ _Float16 smA[2][16384];    // 2 x 32 KB: 8 k-steps x 4 tiles
    const int tid  = threadIdx.x;
    const int wave = tid >> 5;
    const int lane = tid & 31;
    const int m  = lane & 15;
    const int hi = lane >> 4;
    const int rowbase = (blockIdx.x * 8 + wave) * 16;
    const int ks = blockIdx.y;
    const float* xrow = x + (size_t)(rowbase + m) * IN_F;
    const int k_begin = ks * (IN_F / KSPLIT);
    const _Float16* wsrc = wAfrag + (size_t)(k_begin >> 5) * 2048;

    v8f acc0 = {}, acc1 = {}, acc2 = {}, acc3 = {};

    stage32k(wsrc, smA[0], tid);          // prologue: chunk 0
    wait_async0();
    __syncthreads();

    for (int c = 0; c < 4; ++c) {         // 4 chunks x 8 k-steps x 32 K
        if (c + 1 < 4)                    // overlap: stage next chunk
            stage32k(wsrc + (size_t)(c + 1) * 16384, smA[(c + 1) & 1], tid);

        const _Float16* buf = smA[c & 1];
        #pragma unroll
        for (int t = 0; t < 8; ++t) {
            const int k0 = k_begin + (c * 8 + t) * 32;
            // A fragment: two contiguous 32B NT streams per lane, f32 -> f16
            const v4f* p0 = (const v4f*)(xrow + k0 + 8 * hi);
            const v4f* p1 = (const v4f*)(xrow + k0 + 16 + 8 * hi);
            v4f f0 = __builtin_nontemporal_load(p0);
            v4f f1 = __builtin_nontemporal_load(p0 + 1);
            v4f f2 = __builtin_nontemporal_load(p1);
            v4f f3 = __builtin_nontemporal_load(p1 + 1);
            v16h a;
            #pragma unroll
            for (int j = 0; j < 4; ++j) {
                a[j]      = (_Float16)f0[j];
                a[4 + j]  = (_Float16)f1[j];
                a[8 + j]  = (_Float16)f2[j];
                a[12 + j] = (_Float16)f3[j];
            }
            // B fragments from LDS (ds_load_b128 pairs)
            const _Float16* bb = buf + (t * 4) * 512 + lane * 16;
            v16h b0 = *(const v16h*)(bb + 0 * 512);
            v16h b1 = *(const v16h*)(bb + 1 * 512);
            v16h b2 = *(const v16h*)(bb + 2 * 512);
            v16h b3 = *(const v16h*)(bb + 3 * 512);
            acc0 = __builtin_amdgcn_wmma_f32_16x16x32_f16(false, a, false, b0, (short)0, acc0, false, false);
            acc1 = __builtin_amdgcn_wmma_f32_16x16x32_f16(false, a, false, b1, (short)0, acc1, false, false);
            acc2 = __builtin_amdgcn_wmma_f32_16x16x32_f16(false, a, false, b2, (short)0, acc2, false, false);
            acc3 = __builtin_amdgcn_wmma_f32_16x16x32_f16(false, a, false, b3, (short)0, acc3, false, false);
        }
        wait_async0();
        __syncthreads();
    }

    float* outp = xpart + ((size_t)ks * N_ROWS + rowbase) * RANK;
    #pragma unroll
    for (int v = 0; v < 8; ++v) {
        float* rp = outp + (size_t)(v + 8 * hi) * RANK + m;
        __builtin_nontemporal_store(acc0[v], rp + 0);
        __builtin_nontemporal_store(acc1[v], rp + 16);
        __builtin_nontemporal_store(acc2[v], rp + 32);
        __builtin_nontemporal_store(acc3[v], rp + 48);
    }
}

// Sum the KSPLIT partials, convert to fp16 x_proj.
__global__ void reduce_xproj_kernel(const float* __restrict__ xpart,
                                    _Float16* __restrict__ xproj16) {
    int i = blockIdx.x * 256 + threadIdx.x;
    if (i >= N_ROWS * RANK) return;
    const size_t stride = (size_t)N_ROWS * RANK;
    float s = __builtin_nontemporal_load(xpart + i)
            + __builtin_nontemporal_load(xpart + i + stride)
            + __builtin_nontemporal_load(xpart + i + 2 * stride)
            + __builtin_nontemporal_load(xpart + i + 3 * stride);
    xproj16[i] = (_Float16)s;
}

// ---------------------------------------------------------------------------
// GEMM2: out[N, 4096] = fp16(x_proj)[N,64] @ wB16^T[64,4096] + bias.
// Block = 8 waves = 128 rows, all sharing one 1024-column quarter
// (blockIdx.y), so wB fragments are staged once per block in LDS.
__global__ void __launch_bounds__(256)
lora_up_kernel(const _Float16* __restrict__ xproj16,
               const _Float16* __restrict__ wBfrag,
               const float* __restrict__ bias,
               float* __restrict__ out) {
    __shared__ _Float16 smB[2][16384];    // 2 x 32 KB: 16 N-tiles per chunk
    const int tid  = threadIdx.x;
    const int wave = tid >> 5;
    const int lane = tid & 31;
    const int m  = lane & 15;
    const int hi = lane >> 4;
    const int rowbase = (blockIdx.x * 8 + wave) * 16;
    const int q = blockIdx.y;             // column quarter: 64 N-tiles
    const _Float16* wsrc = wBfrag + (size_t)q * 64 * 1024;

    // A fragments (K = 0..63): contiguous 8-half runs per A-frag mapping.
    const _Float16* arow = xproj16 + (size_t)(rowbase + m) * RANK;
    v8h l0 = *(const v8h*)(arow + 8 * hi);
    v8h h0 = *(const v8h*)(arow + 16 + 8 * hi);
    v8h l1 = *(const v8h*)(arow + 32 + 8 * hi);
    v8h h1 = *(const v8h*)(arow + 48 + 8 * hi);
    v16h a0, a1;
    #pragma unroll
    for (int j = 0; j < 8; ++j) {
        a0[j] = l0[j]; a0[8 + j] = h0[j];
        a1[j] = l1[j]; a1[8 + j] = h1[j];
    }

    stage32k(wsrc, smB[0], tid);          // prologue: chunk 0
    wait_async0();
    __syncthreads();

    for (int c = 0; c < 4; ++c) {         // 4 chunks x 16 N-tiles
        if (c + 1 < 4)
            stage32k(wsrc + (size_t)(c + 1) * 16384, smB[(c + 1) & 1], tid);

        const _Float16* buf = smB[c & 1];
        #pragma unroll 4
        for (int lnt = 0; lnt < 16; ++lnt) {
            const int nt = q * 64 + c * 16 + lnt;
            const _Float16* bb = buf + lnt * 1024 + lane * 16;
            v16h b0 = *(const v16h*)(bb);
            v16h b1 = *(const v16h*)(bb + 512);
            v8f cacc = {};
            cacc = __builtin_amdgcn_wmma_f32_16x16x32_f16(false, a0, false, b0, (short)0, cacc, false, false);
            cacc = __builtin_amdgcn_wmma_f32_16x16x32_f16(false, a1, false, b1, (short)0, cacc, false, false);
            float bv = bias[nt * 16 + m];
            float* op = out + (size_t)rowbase * OUT_F + nt * 16 + m;
            #pragma unroll
            for (int v = 0; v < 8; ++v)
                __builtin_nontemporal_store(cacc[v] + bv,
                                            op + (size_t)(v + 8 * hi) * OUT_F);
        }
        wait_async0();
        __syncthreads();
    }
}

extern "C" void kernel_launch(void* const* d_in, const int* in_sizes, int n_in,
                              void* d_out, int out_size, void* d_ws, size_t ws_size,
                              hipStream_t stream) {
    const float* x       = (const float*)d_in[0];
    const float* wA      = (const float*)d_in[1];
    const float* wB      = (const float*)d_in[2];
    const float* bias    = (const float*)d_in[3];
    const float* scaleA  = (const float*)d_in[4];
    const float* scaleB  = (const float*)d_in[5];
    float* out           = (float*)d_out;

    // workspace layout
    char* ws = (char*)d_ws;
    _Float16* wAfrag  = (_Float16*)(ws);                         // 512 KB
    _Float16* wBfrag  = (_Float16*)(ws + (512 << 10));           // 512 KB
    _Float16* xproj16 = (_Float16*)(ws + (1024 << 10));          // 2 MB
    float*    xpart   = (float*)   (ws + (3072 << 10));          // 16 MB

    prep_wA_kernel<<<(IN_F * (RANK / 4)) / 256, 256, 0, stream>>>(wA, scaleA, wAfrag);
    prep_wB_kernel<<<(OUT_F * (RANK / 4)) / 256, 256, 0, stream>>>(wB, scaleB, wBfrag);

    dim3 g1(N_ROWS / 128, KSPLIT);
    lora_down_kernel<<<g1, 256, 0, stream>>>(x, wAfrag, xpart);

    reduce_xproj_kernel<<<(N_ROWS * RANK) / 256, 256, 0, stream>>>(xpart, xproj16);

    dim3 g2(N_ROWS / 128, 4);
    lora_up_kernel<<<g2, 256, 0, stream>>>(xproj16, wBfrag, bias, out);
}
